// Net_12223476924670
// MI455X (gfx1250) — matrix operand order repaired and checked by reading, not compiled, not run on hardware
//
#include <hip/hip_runtime.h>

typedef __attribute__((ext_vector_type(16))) _Float16 v16h;
typedef __attribute__((ext_vector_type(8)))  _Float16 v8h;
typedef __attribute__((ext_vector_type(8)))  float    v8f;

namespace {

constexpr int B_  = 16;
constexpr int N0  = 2048;   // points per cloud
constexpr int M1  = 1024;   // after FPS1
constexpr int M2  = 256;    // after FPS2
constexpr int KN  = 64;     // max neighbors

// ---------------------------------------------------------------- transpose
// data [B,3,N] -> pos [B,N,3]
__global__ void transpose_pos_kernel(const float* __restrict__ data,
                                     float* __restrict__ pos) {
  int t = blockIdx.x * blockDim.x + threadIdx.x;
  if (t >= B_ * N0) return;
  int b = t >> 11, n = t & (N0 - 1);
  const float* d = data + (size_t)b * 3 * N0;
  float* p = pos + (size_t)t * 3;
  p[0] = d[n]; p[1] = d[N0 + n]; p[2] = d[2 * N0 + n];
}

// ---------------------------------------------------------------- FPS
// One block (256 thr) per cloud. Deterministic start at index 0, argmax with
// first-occurrence (lowest index) tie-break to match jnp.argmax.
template <int NPTS, int NSAMP>
__global__ void fps_kernel(const float* __restrict__ pos,   // [B][NPTS][3]
                           int* __restrict__ idx,           // [B][NSAMP]
                           float* __restrict__ posd) {      // [B][NSAMP][3]
  int b = blockIdx.x;
  const float* P = pos + (size_t)b * NPTS * 3;
  __shared__ float dists[NPTS];
  __shared__ float rv[256];
  __shared__ int   ri[256];
  __shared__ float lp[3];
  int t = threadIdx.x;
  for (int i = t; i < NPTS; i += 256) dists[i] = 1e9f;
  int last = 0;
  for (int s = 0; s < NSAMP; ++s) {
    if (t == 0) {
      idx[b * NSAMP + s] = last;
      lp[0] = P[last * 3 + 0]; lp[1] = P[last * 3 + 1]; lp[2] = P[last * 3 + 2];
      posd[((size_t)b * NSAMP + s) * 3 + 0] = lp[0];
      posd[((size_t)b * NSAMP + s) * 3 + 1] = lp[1];
      posd[((size_t)b * NSAMP + s) * 3 + 2] = lp[2];
    }
    __syncthreads();
    float lx = lp[0], ly = lp[1], lz = lp[2];
    float bestv = -1.0f; int besti = 0x7fffffff;
    for (int i = t; i < NPTS; i += 256) {      // each thread owns i % 256 == t
      float dx = P[i * 3 + 0] - lx;
      float dy = P[i * 3 + 1] - ly;
      float dz = P[i * 3 + 2] - lz;
      float d = dx * dx + dy * dy + dz * dz;
      float nd = fminf(dists[i], d);
      dists[i] = nd;
      if (nd > bestv) { bestv = nd; besti = i; }   // ascending i -> first max kept
    }
    rv[t] = bestv; ri[t] = besti;
    __syncthreads();
    for (int off = 128; off > 0; off >>= 1) {
      if (t < off) {
        if (rv[t + off] > rv[t] ||
            (rv[t + off] == rv[t] && ri[t + off] < ri[t])) {
          rv[t] = rv[t + off]; ri[t] = ri[t + off];
        }
      }
      __syncthreads();
    }
    last = ri[0];
    __syncthreads();
  }
}

// ---------------------------------------------------------------- radius kNN
// One wave32 per destination point. Exact k-nearest-within-radius by selecting
// in strictly increasing (d2, idx) lexicographic order (set-equivalent to
// lax.top_k since aggregation is max). Valid entries come first => cnt mask.
template <int NSRC>
__global__ __launch_bounds__(32)
void radius_topk_kernel(const float* __restrict__ src,   // [B][NSRC][3]
                        const float* __restrict__ dst,   // [B][mdst][3]
                        int mdst, float r2,
                        int* __restrict__ nbr,           // [B*mdst][64]
                        int* __restrict__ cnt) {         // [B*mdst]
  int g = blockIdx.x;                 // b*mdst + m
  int b = g / mdst;
  int lane = threadIdx.x;
  __shared__ float d2s[NSRC];
  const float* S = src + (size_t)b * NSRC * 3;
  const float* D = dst + (size_t)g * 3;
  float x0 = D[0], y0 = D[1], z0 = D[2];
  for (int i = lane; i < NSRC; i += 32) {
    float dx = S[i * 3 + 0] - x0;
    float dy = S[i * 3 + 1] - y0;
    float dz = S[i * 3 + 2] - z0;
    d2s[i] = dx * dx + dy * dy + dz * dz;
  }
  __syncthreads();
  float pv = -1.0f; int pi = -1;
  int c = 0;
  for (int j = 0; j < KN; ++j) {
    float bv = 3e38f; int bi = 0x7fffffff;
    for (int i = lane; i < NSRC; i += 32) {
      float v = d2s[i];
      if (v > pv || (v == pv && i > pi)) {        // strictly after previous pick
        if (v < bv || (v == bv && i < bi)) { bv = v; bi = i; }
      }
    }
    for (int off = 16; off > 0; off >>= 1) {
      float ov = __shfl_xor(bv, off, 32);
      int   oi = __shfl_xor(bi, off, 32);
      if (ov < bv || (ov == bv && oi < bi)) { bv = ov; bi = oi; }
    }
    pv = bv; pi = bi;
    bool valid = (bi != 0x7fffffff) && (bv <= r2);
    if (lane == 0) nbr[(size_t)g * KN + j] = valid ? bi : 0;
    if (valid) ++c;
  }
  if (lane == 0) cnt[g] = c;
}

// ---------------------------------------------------------------- gathers
// SA1 edge features: rel = pos[nbr]-pos_dst, padded 3 -> 32 (zeros)
__global__ void gather_feat1_kernel(const float* __restrict__ src,   // [N0][3]
                                    const float* __restrict__ dstp,  // [M1][3]
                                    const int* __restrict__ nbr,     // [M1][64]
                                    const int* __restrict__ cnt,     // [M1]
                                    _Float16* __restrict__ feat) {   // [M1*64][32]
  int row = blockIdx.x * blockDim.x + threadIdx.x;
  if (row >= M1 * KN) return;
  int m = row >> 6, k = row & (KN - 1);
  float f0 = 0.f, f1 = 0.f, f2 = 0.f;
  if (k < cnt[m]) {
    int j = nbr[row];
    f0 = src[j * 3 + 0] - dstp[m * 3 + 0];
    f1 = src[j * 3 + 1] - dstp[m * 3 + 1];
    f2 = src[j * 3 + 2] - dstp[m * 3 + 2];
  }
  _Float16* o = feat + (size_t)row * 32;
  o[0] = (_Float16)f0; o[1] = (_Float16)f1; o[2] = (_Float16)f2;
  for (int i = 3; i < 32; ++i) o[i] = (_Float16)0.f;
}

// SA2 edge features: [x1[nbr](128) | rel(3) | pad(29)] -> 160
__global__ void gather_feat2_kernel(const _Float16* __restrict__ x1,  // [M1][128]
                                    const float* __restrict__ src,    // [M1][3]
                                    const float* __restrict__ dstp,   // [M2][3]
                                    const int* __restrict__ nbr,      // [M2][64]
                                    const int* __restrict__ cnt,      // [M2]
                                    _Float16* __restrict__ feat) {    // [M2*64][160]
  int row = blockIdx.x * blockDim.x + threadIdx.x;
  if (row >= M2 * KN) return;
  int m = row >> 6, k = row & (KN - 1);
  _Float16* o = feat + (size_t)row * 160;
  if (k < cnt[m]) {
    int j = nbr[row];
    const _Float16* xi = x1 + (size_t)j * 128;
    for (int i = 0; i < 128; ++i) o[i] = xi[i];
    o[128] = (_Float16)(src[j * 3 + 0] - dstp[m * 3 + 0]);
    o[129] = (_Float16)(src[j * 3 + 1] - dstp[m * 3 + 1]);
    o[130] = (_Float16)(src[j * 3 + 2] - dstp[m * 3 + 2]);
  } else {
    for (int i = 0; i < 131; ++i) o[i] = (_Float16)0.f;
  }
  for (int i = 131; i < 160; ++i) o[i] = (_Float16)0.f;
}

// SA3 features (batched): [x2(256) | pos2(3) | pad(29)] -> 288
__global__ void gather_feat3_kernel(const _Float16* __restrict__ x2,  // [B*M2][256]
                                    const float* __restrict__ pos2,   // [B*M2][3]
                                    _Float16* __restrict__ feat) {    // [B*M2][288]
  int row = blockIdx.x * blockDim.x + threadIdx.x;
  if (row >= B_ * M2) return;
  _Float16* o = feat + (size_t)row * 288;
  const _Float16* xi = x2 + (size_t)row * 256;
  for (int i = 0; i < 256; ++i) o[i] = xi[i];
  o[256] = (_Float16)pos2[row * 3 + 0];
  o[257] = (_Float16)pos2[row * 3 + 1];
  o[258] = (_Float16)pos2[row * 3 + 2];
  for (int i = 259; i < 288; ++i) o[i] = (_Float16)0.f;
}

// ---------------------------------------------------------------- WMMA GEMM
// C[M,N](f16) = act(A[M,Kdp](f16) @ Wt^T + bias),  Wt stored [N][Kdp].
// Block = 8 waves = 64 rows x 32 cols of C; one 16x16 WMMA tile per wave.
template <bool RELU>
__global__ __launch_bounds__(256)
void gemm_wmma_kernel(const _Float16* __restrict__ A, int lda,
                      const _Float16* __restrict__ Wt,          // [N][Kdp]
                      const float* __restrict__ bias,
                      _Float16* __restrict__ C, int ldc,
                      int M, int N, int Kdp) {
  int lane = threadIdx.x & 31;
  int wave = threadIdx.x >> 5;           // 0..7
  int wm = wave & 3, wn = wave >> 2;     // 4 x 2 tile grid
  int row0 = blockIdx.x * 64 + wm * 16;
  int col0 = blockIdx.y * 32 + wn * 16;
  if (row0 >= M || col0 >= N) return;    // wave-uniform
  int sel = lane & 15;                   // A row / B col within tile
  int kh  = lane >> 4;                   // K-half select
  const _Float16* arow = A  + (size_t)(row0 + sel) * lda;
  const _Float16* brow = Wt + (size_t)(col0 + sel) * Kdp;
  v8f acc = {};
  for (int kg = 0; kg < Kdp; kg += 32) {
    v8h alo = *(const v8h*)(arow + kg + kh * 8);        // K = kh*8 + [0..7]
    v8h ahi = *(const v8h*)(arow + kg + 16 + kh * 8);   // K = 16 + kh*8 + [0..7]
    v16h av = __builtin_shufflevector(alo, ahi,
        0, 1, 2, 3, 4, 5, 6, 7, 8, 9, 10, 11, 12, 13, 14, 15);
    v8h blo = *(const v8h*)(brow + kg + kh * 8);        // K = kh*8 + [0..7]
    v8h bhi = *(const v8h*)(brow + kg + 16 + kh * 8);   // K = 16 + kh*8 + [0..7]
    v16h bv = __builtin_shufflevector(blo, bhi,
        0, 1, 2, 3, 4, 5, 6, 7, 8, 9, 10, 11, 12, 13, 14, 15);
    acc = __builtin_amdgcn_wmma_f32_16x16x32_f16(
        false, av, false, bv, (short)0, acc, false, false);
  }
  int n = col0 + sel;
  float bi = bias[n];
#pragma unroll
  for (int v = 0; v < 8; ++v) {
    int m = row0 + v + 8 * kh;           // C/D layout: VGPR v -> row v + 8*kh
    float x = acc[v] + bi;
    if (RELU) x = fmaxf(x, 0.f);
    C[(size_t)m * ldc + n] = (_Float16)x;
  }
}

// ---------------------------------------------------------------- max-aggr
__global__ void aggr_max_kernel(const _Float16* __restrict__ H, int ldh, // [mdst*64][C]
                                const int* __restrict__ cnt,
                                _Float16* __restrict__ X,                // [mdst][C]
                                int mdst, int C) {
  int t = blockIdx.x * blockDim.x + threadIdx.x;
  if (t >= mdst * C) return;
  int m = t / C, c = t % C;
  int nv = cnt[m];
  float best = -1e9f;
  for (int k = 0; k < nv; ++k)
    best = fmaxf(best, (float)H[(size_t)(m * KN + k) * ldh + c]);
  X[(size_t)m * C + c] = (_Float16)(nv > 0 ? best : 0.f);
}

// global max over the 256 points of each cloud; pad rows 16..63 with zeros
__global__ void rowmax_kernel(const _Float16* __restrict__ H,  // [B*M2][1024]
                              _Float16* __restrict__ G) {      // [64][1024]
  int t = blockIdx.x * blockDim.x + threadIdx.x;
  if (t >= 64 * 1024) return;
  int r = t >> 10, c = t & 1023;
  float best = 0.f;
  if (r < B_) {
    best = -3e38f;
    for (int m = 0; m < M2; ++m)
      best = fmaxf(best, (float)H[(size_t)(r * M2 + m) * 1024 + c]);
  }
  G[t] = (_Float16)best;
}

// ---------------------------------------------------------------- head final
__global__ void head_final_kernel(const _Float16* __restrict__ h,  // [>=16][256]
                                  const float* __restrict__ W,     // [256][10]
                                  const float* __restrict__ bias,  // [10]
                                  float* __restrict__ out) {       // [16][10]
  __shared__ float lg[16][10];
  int t = threadIdx.x;
  if (t < 160) {
    int b = t / 10, j = t % 10;
    float s = bias[j];
    const _Float16* hr = h + (size_t)b * 256;
    for (int i = 0; i < 256; ++i) s += (float)hr[i] * W[i * 10 + j];
    lg[b][j] = s;
  }
  __syncthreads();
  if (t < 16) {
    float mx = -3e38f;
    for (int j = 0; j < 10; ++j) mx = fmaxf(mx, lg[t][j]);
    float se = 0.f;
    for (int j = 0; j < 10; ++j) se += expf(lg[t][j] - mx);
    float lse = mx + logf(se);
    for (int j = 0; j < 10; ++j) out[t * 10 + j] = lg[t][j] - lse;
  }
}

// ---------------------------------------------------------------- weights
// W f32 [Cin][Cout] -> Wt f16 [Cout][Kdp] (transposed, K padded with zeros)
__global__ void convert_w_kernel(const float* __restrict__ W,
                                 _Float16* __restrict__ Wt,
                                 int Cin, int Cout, int Kdp) {
  int t = blockIdx.x * blockDim.x + threadIdx.x;
  if (t >= Cout * Kdp) return;
  int n = t / Kdp, k = t % Kdp;
  Wt[t] = (_Float16)(k < Cin ? W[(size_t)k * Cout + n] : 0.f);
}

} // namespace

extern "C" void kernel_launch(void* const* d_in, const int* in_sizes, int n_in,
                              void* d_out, int out_size, void* d_ws, size_t ws_size,
                              hipStream_t stream) {
  (void)in_sizes; (void)n_in; (void)out_size; (void)ws_size;
  const float* data = (const float*)d_in[0];

  char* ws = (char*)d_ws;
  size_t cur = 0;
  auto alloc = [&](size_t bytes) -> char* {
    char* p = ws + cur;
    cur = (cur + bytes + 255) & ~(size_t)255;
    return p;
  };

  // ---- weights: convert/transpose/pad to f16 once per launch (deterministic)
  struct Layer { int wi, bi, cin, cout, kdp; _Float16* wt; };
  Layer L[11] = {
      {1, 2, 3, 64, 32, nullptr},   {3, 4, 64, 64, 64, nullptr},   {5, 6, 64, 128, 64, nullptr},
      {7, 8, 131, 128, 160, nullptr}, {9, 10, 128, 128, 128, nullptr}, {11, 12, 128, 256, 128, nullptr},
      {13, 14, 259, 256, 288, nullptr}, {15, 16, 256, 512, 256, nullptr}, {17, 18, 512, 1024, 512, nullptr},
      {19, 20, 1024, 512, 1024, nullptr}, {21, 22, 512, 256, 512, nullptr}};
  for (int i = 0; i < 11; ++i) {
    L[i].wt = (_Float16*)alloc((size_t)L[i].cout * L[i].kdp * 2);
    int tot = L[i].cout * L[i].kdp;
    convert_w_kernel<<<(tot + 255) / 256, 256, 0, stream>>>(
        (const float*)d_in[L[i].wi], L[i].wt, L[i].cin, L[i].cout, L[i].kdp);
  }
  auto BIAS = [&](int li) { return (const float*)d_in[L[li].bi]; };

  // ---- persistent buffers
  float* pos0 = (float*)alloc((size_t)B_ * N0 * 3 * 4);
  int*   idx1 = (int*)alloc((size_t)B_ * M1 * 4);
  float* pos1 = (float*)alloc((size_t)B_ * M1 * 3 * 4);
  int*   idx2 = (int*)alloc((size_t)B_ * M2 * 4);
  float* pos2 = (float*)alloc((size_t)B_ * M2 * 3 * 4);
  int*   nbr1 = (int*)alloc((size_t)B_ * M1 * KN * 4);
  int*   cnt1 = (int*)alloc((size_t)B_ * M1 * 4);
  int*   nbr2 = (int*)alloc((size_t)B_ * M2 * KN * 4);
  int*   cnt2 = (int*)alloc((size_t)B_ * M2 * 4);
  _Float16* x1 = (_Float16*)alloc((size_t)B_ * M1 * 128 * 2);
  _Float16* x2 = (_Float16*)alloc((size_t)B_ * M2 * 256 * 2);
  _Float16* feat3 = (_Float16*)alloc((size_t)B_ * M2 * 288 * 2);
  _Float16* gx  = (_Float16*)alloc((size_t)64 * 1024 * 2);
  _Float16* hh1 = (_Float16*)alloc((size_t)64 * 512 * 2);
  _Float16* hh2 = (_Float16*)alloc((size_t)64 * 256 * 2);
  _Float16* P0 = (_Float16*)alloc((size_t)8  * 1024 * 1024);   // ping
  _Float16* P1 = (_Float16*)alloc((size_t)16 * 1024 * 1024);   // pong

  // ---- geometry stage (batched): transpose, FPS1, FPS2, radius graphs
  transpose_pos_kernel<<<(B_ * N0 + 255) / 256, 256, 0, stream>>>(data, pos0);
  fps_kernel<N0, M1><<<B_, 256, 0, stream>>>(pos0, idx1, pos1);
  fps_kernel<M1, M2><<<B_, 256, 0, stream>>>(pos1, idx2, pos2);
  radius_topk_kernel<N0><<<B_ * M1, 32, 0, stream>>>(pos0, pos1, M1, 0.2f * 0.2f, nbr1, cnt1);
  radius_topk_kernel<M1><<<B_ * M2, 32, 0, stream>>>(pos1, pos2, M2, 0.4f * 0.4f, nbr2, cnt2);

  // ---- SA1 + SA2 heavy stage: per-cloud to cap scratch (P0/P1 ping-pong)
  for (int b = 0; b < B_; ++b) {
    // SA1: feat[65536][32] -> 64 -> 64 -> 128, max over K
    gather_feat1_kernel<<<(M1 * KN + 255) / 256, 256, 0, stream>>>(
        pos0 + (size_t)b * N0 * 3, pos1 + (size_t)b * M1 * 3,
        nbr1 + (size_t)b * M1 * KN, cnt1 + (size_t)b * M1, P0);
    gemm_wmma_kernel<true><<<dim3(M1 * KN / 64, 64 / 32), 256, 0, stream>>>(
        P0, 32, L[0].wt, BIAS(0), P1, 64, M1 * KN, 64, 32);
    gemm_wmma_kernel<true><<<dim3(M1 * KN / 64, 64 / 32), 256, 0, stream>>>(
        P1, 64, L[1].wt, BIAS(1), P0, 64, M1 * KN, 64, 64);
    gemm_wmma_kernel<false><<<dim3(M1 * KN / 64, 128 / 32), 256, 0, stream>>>(
        P0, 64, L[2].wt, BIAS(2), P1, 128, M1 * KN, 128, 64);
    aggr_max_kernel<<<(M1 * 128 + 255) / 256, 256, 0, stream>>>(
        P1, 128, cnt1 + (size_t)b * M1, x1 + (size_t)b * M1 * 128, M1, 128);

    // SA2: feat[16384][160] -> 128 -> 128 -> 256, max over K
    gather_feat2_kernel<<<(M2 * KN + 255) / 256, 256, 0, stream>>>(
        x1 + (size_t)b * M1 * 128, pos1 + (size_t)b * M1 * 3,
        pos2 + (size_t)b * M2 * 3, nbr2 + (size_t)b * M2 * KN,
        cnt2 + (size_t)b * M2, P0);
    gemm_wmma_kernel<true><<<dim3(M2 * KN / 64, 128 / 32), 256, 0, stream>>>(
        P0, 160, L[3].wt, BIAS(3), P1, 128, M2 * KN, 128, 160);
    gemm_wmma_kernel<true><<<dim3(M2 * KN / 64, 128 / 32), 256, 0, stream>>>(
        P1, 128, L[4].wt, BIAS(4), P0, 128, M2 * KN, 128, 128);
    gemm_wmma_kernel<false><<<dim3(M2 * KN / 64, 256 / 32), 256, 0, stream>>>(
        P0, 128, L[5].wt, BIAS(5), P1, 256, M2 * KN, 256, 128);
    aggr_max_kernel<<<(M2 * 256 + 255) / 256, 256, 0, stream>>>(
        P1, 256, cnt2 + (size_t)b * M2, x2 + (size_t)b * M2 * 256, M2, 256);
  }

  // ---- SA3 (batched, M = B*256 = 4096): 288 -> 256 -> 512 -> 1024, global max
  gather_feat3_kernel<<<(B_ * M2 + 255) / 256, 256, 0, stream>>>(x2, pos2, feat3);
  gemm_wmma_kernel<true><<<dim3(B_ * M2 / 64, 256 / 32), 256, 0, stream>>>(
      feat3, 288, L[6].wt, BIAS(6), P1, 256, B_ * M2, 256, 288);
  gemm_wmma_kernel<true><<<dim3(B_ * M2 / 64, 512 / 32), 256, 0, stream>>>(
      P1, 256, L[7].wt, BIAS(7), P0, 512, B_ * M2, 512, 256);
  gemm_wmma_kernel<false><<<dim3(B_ * M2 / 64, 1024 / 32), 256, 0, stream>>>(
      P0, 512, L[8].wt, BIAS(8), P1, 1024, B_ * M2, 1024, 512);
  rowmax_kernel<<<(64 * 1024 + 255) / 256, 256, 0, stream>>>(P1, gx);

  // ---- head: [16(pad 64),1024] -> 512 -> 256 -> 10, log_softmax
  gemm_wmma_kernel<true><<<dim3(1, 512 / 32), 256, 0, stream>>>(
      gx, 1024, L[9].wt, BIAS(9), hh1, 512, 64, 512, 1024);
  gemm_wmma_kernel<true><<<dim3(1, 256 / 32), 256, 0, stream>>>(
      hh1, 512, L[10].wt, BIAS(10), hh2, 256, 64, 256, 512);
  head_final_kernel<<<1, 256, 0, stream>>>(
      hh2, (const float*)d_in[23], (const float*)d_in[24], (float*)d_out);
}
